// AttnDecoderRNN_80719615361528
// MI455X (gfx1250) — matrix operand order repaired and checked by reading, not compiled, not run on hardware
//
#include <hip/hip_runtime.h>
#include <math.h>

// ---- problem constants ----
#define H   512
#define V   16000
#define T   100
#define B   32
#define S   64
#define G3  1536   // 3*H
#define H2  1024   // 2*H

// ---- WMMA vector types (gfx1250, wave32) ----
typedef __attribute__((ext_vector_type(16))) __bf16 v16bf;
typedef __attribute__((ext_vector_type(8)))  __bf16 v8bf;
typedef __attribute__((ext_vector_type(8)))  float  v8f;

__device__ __forceinline__ unsigned short f32_to_bf16(float f) {
  unsigned int u = __float_as_uint(f);
  unsigned int r = 0x7FFFu + ((u >> 16) & 1u);
  return (unsigned short)((u + r) >> 16);
}
__device__ __forceinline__ float sigmoidf(float x) { return 1.f / (1.f + expf(-x)); }

// A fragment (16-bit A, 16x32): lane<16 holds K {0..7,16..23}, lane>=16 holds {8..15,24..31}.
// p must already include the per-lane half offset (half*8 elements).
__device__ __forceinline__ v16bf load_a(const unsigned short* p) {
  v8bf lo = *(const v8bf*)(p);
  v8bf hi = *(const v8bf*)(p + 16);
  v16bf a;
#pragma unroll
  for (int i = 0; i < 8; ++i) { a[i] = lo[i]; a[i + 8] = hi[i]; }
  return a;
}
// B fragment (32x16): lane<16 holds K {0..7,16..23}, lane>=16 holds {8..15,24..31}.
// p must already include the per-lane half offset (half*8 elements): two 16B loads.
__device__ __forceinline__ v16bf load_b(const unsigned short* p) {
  v8bf lo = *(const v8bf*)(p);
  v8bf hi = *(const v8bf*)(p + 16);
  v16bf b;
#pragma unroll
  for (int i = 0; i < 8; ++i) { b[i] = lo[i]; b[i + 8] = hi[i]; }
  return b;
}
#define WMMA_BF16(Afrag, Bfrag, Cacc) \
  __builtin_amdgcn_wmma_f32_16x16x32_bf16(false, (Afrag), false, (Bfrag), (short)0, (Cacc), false, false)

// ---------------- one-time kernels ----------------
__global__ void convert_bf16_kernel(const float* __restrict__ src,
                                    unsigned short* __restrict__ dst, int n) {
  int i = blockIdx.x * blockDim.x + threadIdx.x;
  if (i < n) dst[i] = f32_to_bf16(src[i]);
}

__global__ void init_kernel(const float* __restrict__ eh, float* __restrict__ h,
                            unsigned short* __restrict__ hb, int* __restrict__ ids) {
  int i = blockIdx.x * blockDim.x + threadIdx.x;
  if (i < B * H) { float v = eh[i]; h[i] = v; hb[i] = f32_to_bf16(v); }
  if (i < B) ids[i] = 0;
}

// Uk = keys @ Ua_w.T + Ua_b : M=B*S=2048, N=H, K=H  (WMMA bf16, f32 acc)
__global__ __launch_bounds__(256)
void uk_gemm_kernel(const unsigned short* __restrict__ keysb,
                    const unsigned short* __restrict__ Uab,
                    const float* __restrict__ Ua_bias, float* __restrict__ Uk) {
  int wave = threadIdx.x >> 5, lane = threadIdx.x & 31;
  int r = lane & 15, half = lane >> 4;
  int tile = blockIdx.x * 8 + wave;          // 128 Mtiles * 32 Ntiles = 4096
  int m0 = (tile >> 5) << 4;
  int n  = ((tile & 31) << 4) + r;
  const unsigned short* a  = keysb + (size_t)(m0 + r) * H + half * 8;
  const unsigned short* bc = Uab   + (size_t)n * H + half * 8;
  v8f acc = {};
  for (int k = 0; k < H; k += 32)
    acc = WMMA_BF16(load_a(a + k), load_b(bc + k), acc);
  float bias = Ua_bias[n];
  int mb = half * 8;
#pragma unroll
  for (int i = 0; i < 8; ++i)
    Uk[(size_t)(m0 + mb + i) * H + n] = acc[i] + bias;
}

// ---------------- per-step kernels ----------------
// Attention: query GEMV, e=tanh(q+Uk), scores->softmax, ctx; emits x=[emb,ctx] in bf16.
__global__ __launch_bounds__(256)
void attn_kernel(const float* __restrict__ h, const int* __restrict__ ids,
                 const float* __restrict__ embedding, const float* __restrict__ keys,
                 const float* __restrict__ Uk, const float* __restrict__ Wa_w,
                 const float* __restrict__ Wa_b, const float* __restrict__ Va_w,
                 const float* __restrict__ Va_b, unsigned short* __restrict__ xb,
                 float* __restrict__ attn_out, int t) {
  int b = blockIdx.x, tid = threadIdx.x;
  __shared__ float hs[H], q[H], part[256], w[S];
  __shared__ float s_red;
  const float* hrow = h + (size_t)b * H;
  for (int i = tid; i < H; i += 256) hs[i] = hrow[i];
  __syncthreads();
  for (int j = tid; j < H; j += 256) {
    const float* wr = Wa_w + (size_t)j * H;
    float acc = 0.f;
    for (int k = 0; k < H; ++k) acc = fmaf(hs[k], wr[k], acc);
    q[j] = acc + Wa_b[j];
  }
  __syncthreads();
  {
    int s = tid >> 2, l4 = tid & 3;
    const float* ukr = Uk + ((size_t)b * S + s) * H;
    float p = 0.f;
    for (int k = l4; k < H; k += 4) p = fmaf(tanhf(q[k] + ukr[k]), Va_w[k], p);
    part[tid] = p;
  }
  __syncthreads();
  if (tid < S) w[tid] = part[4 * tid] + part[4 * tid + 1] + part[4 * tid + 2] + part[4 * tid + 3] + Va_b[0];
  __syncthreads();
  if (tid == 0) { float m = w[0]; for (int s = 1; s < S; ++s) m = fmaxf(m, w[s]); s_red = m; }
  __syncthreads();
  if (tid < S) w[tid] = expf(w[tid] - s_red);
  __syncthreads();
  if (tid == 0) { float s = 0.f; for (int i = 0; i < S; ++i) s += w[i]; s_red = 1.f / s; }
  __syncthreads();
  if (tid < S) {
    w[tid] *= s_red;
    attn_out[((size_t)b * T + t) * S + tid] = w[tid];
  }
  __syncthreads();
  const float* krow = keys + (size_t)b * S * H;
  for (int j = tid; j < H; j += 256) {
    float c = 0.f;
    for (int s = 0; s < S; ++s) c = fmaf(w[s], krow[(size_t)s * H + j], c);
    xb[(size_t)b * H2 + H + j] = f32_to_bf16(c);
  }
  int id = ids[b];
  const float* erow = embedding + (size_t)id * H;
  for (int j = tid; j < H; j += 256) xb[(size_t)b * H2 + j] = f32_to_bf16(erow[j]);
}

// gx = x@W_ih.T + b_ih ; gh = h@W_hh.T + b_hh  (kept separate for the GRU n-gate)
__global__ __launch_bounds__(256)
void gates_gemm_kernel(const unsigned short* __restrict__ xb,
                       const unsigned short* __restrict__ hb,
                       const unsigned short* __restrict__ Wihb,
                       const unsigned short* __restrict__ Whhb,
                       const float* __restrict__ b_ih, const float* __restrict__ b_hh,
                       float* __restrict__ gx, float* __restrict__ gh) {
  int wave = threadIdx.x >> 5, lane = threadIdx.x & 31;
  int r = lane & 15, half = lane >> 4;
  int n = (blockIdx.x * 8 + wave) * 16 + r;   // 96 N-tiles total
  int mb = half * 8;
  {
    const unsigned short* a0 = xb + (size_t)r * H2 + half * 8;
    const unsigned short* a1 = xb + (size_t)(16 + r) * H2 + half * 8;
    const unsigned short* bc = Wihb + (size_t)n * H2 + half * 8;
    v8f acc0 = {}, acc1 = {};
    for (int k = 0; k < H2; k += 32) {
      v16bf bf = load_b(bc + k);
      acc0 = WMMA_BF16(load_a(a0 + k), bf, acc0);
      acc1 = WMMA_BF16(load_a(a1 + k), bf, acc1);
    }
    float bias = b_ih[n];
#pragma unroll
    for (int i = 0; i < 8; ++i) {
      gx[(size_t)(mb + i) * G3 + n]      = acc0[i] + bias;
      gx[(size_t)(16 + mb + i) * G3 + n] = acc1[i] + bias;
    }
  }
  {
    const unsigned short* a0 = hb + (size_t)r * H + half * 8;
    const unsigned short* a1 = hb + (size_t)(16 + r) * H + half * 8;
    const unsigned short* bc = Whhb + (size_t)n * H + half * 8;
    v8f acc0 = {}, acc1 = {};
    for (int k = 0; k < H; k += 32) {
      v16bf bf = load_b(bc + k);
      acc0 = WMMA_BF16(load_a(a0 + k), bf, acc0);
      acc1 = WMMA_BF16(load_a(a1 + k), bf, acc1);
    }
    float bias = b_hh[n];
#pragma unroll
    for (int i = 0; i < 8; ++i) {
      gh[(size_t)(mb + i) * G3 + n]      = acc0[i] + bias;
      gh[(size_t)(16 + mb + i) * G3 + n] = acc1[i] + bias;
    }
  }
}

__global__ void gru_elem_kernel(const float* __restrict__ gx, const float* __restrict__ gh,
                                float* __restrict__ h, unsigned short* __restrict__ hb,
                                float* __restrict__ hT_out, int write_hT) {
  int i = blockIdx.x * blockDim.x + threadIdx.x;  // B*H
  int b = i >> 9, j = i & (H - 1);
  const float* gxr = gx + (size_t)b * G3;
  const float* ghr = gh + (size_t)b * G3;
  float r = sigmoidf(gxr[j] + ghr[j]);
  float z = sigmoidf(gxr[H + j] + ghr[H + j]);
  float n = tanhf(gxr[2 * H + j] + r * ghr[2 * H + j]);
  float hv = (1.f - z) * n + z * h[i];
  h[i] = hv;
  hb[i] = f32_to_bf16(hv);
  if (write_hT) hT_out[i] = hv;
}

// logits = h_new @ out_w.T + out_b : M=32, N=16000, K=512
__global__ __launch_bounds__(256)
void logits_gemm_kernel(const unsigned short* __restrict__ hb,
                        const unsigned short* __restrict__ Wb,
                        const float* __restrict__ out_b, float* __restrict__ logits) {
  int wave = threadIdx.x >> 5, lane = threadIdx.x & 31;
  int r = lane & 15, half = lane >> 4;
  int n = (blockIdx.x * 8 + wave) * 16 + r;   // 1000 N-tiles
  const unsigned short* a0 = hb + (size_t)r * H + half * 8;
  const unsigned short* a1 = hb + (size_t)(16 + r) * H + half * 8;
  const unsigned short* bc = Wb + (size_t)n * H + half * 8;
  v8f acc0 = {}, acc1 = {};
  for (int k = 0; k < H; k += 32) {
    v16bf bf = load_b(bc + k);
    acc0 = WMMA_BF16(load_a(a0 + k), bf, acc0);
    acc1 = WMMA_BF16(load_a(a1 + k), bf, acc1);
  }
  float bias = out_b[n];
  int mb = half * 8;
#pragma unroll
  for (int i = 0; i < 8; ++i) {
    logits[(size_t)(mb + i) * V + n]      = acc0[i] + bias;
    logits[(size_t)(16 + mb + i) * V + n] = acc1[i] + bias;
  }
}

// argmax + log_softmax over V, emits ids for next step and log_probs[b,t,:]
__global__ __launch_bounds__(256)
void reduce_kernel(const float* __restrict__ logits, float* __restrict__ log_probs,
                   int* __restrict__ ids, int t) {
  int b = blockIdx.x, tid = threadIdx.x;
  const float* row = logits + (size_t)b * V;
  __shared__ float smax[256]; __shared__ int sarg[256]; __shared__ float ssum[256];
  float m = -INFINITY; int mi = 0;
  for (int v = tid; v < V; v += 256) {
    float x = row[v];
    if (x > m) { m = x; mi = v; }
  }
  smax[tid] = m; sarg[tid] = mi; __syncthreads();
  for (int st = 128; st > 0; st >>= 1) {
    if (tid < st) {
      float om = smax[tid + st]; int oi = sarg[tid + st];
      if (om > smax[tid] || (om == smax[tid] && oi < sarg[tid])) { smax[tid] = om; sarg[tid] = oi; }
    }
    __syncthreads();
  }
  float gmax = smax[0];
  if (tid == 0) ids[b] = sarg[0];
  float s = 0.f;
  for (int v = tid; v < V; v += 256) s += expf(row[v] - gmax);
  ssum[tid] = s; __syncthreads();
  for (int st = 128; st > 0; st >>= 1) { if (tid < st) ssum[tid] += ssum[tid + st]; __syncthreads(); }
  float logZ = gmax + logf(ssum[0]);
  float* dst = log_probs + ((size_t)b * T + t) * (size_t)V;
  for (int v = tid; v < V; v += 256) dst[v] = row[v] - logZ;
}

// ---------------- host launcher ----------------
extern "C" void kernel_launch(void* const* d_in, const int* in_sizes, int n_in,
                              void* d_out, int out_size, void* d_ws, size_t ws_size,
                              hipStream_t stream) {
  const float* encoder_outputs = (const float*)d_in[0];   // [B,S,H]
  const float* encoder_hidden  = (const float*)d_in[1];   // [1,B,H]
  const float* embedding       = (const float*)d_in[2];   // [V,H]
  const float* Wa_w = (const float*)d_in[3];
  const float* Wa_b = (const float*)d_in[4];
  const float* Ua_w = (const float*)d_in[5];
  const float* Ua_b = (const float*)d_in[6];
  const float* Va_w = (const float*)d_in[7];
  const float* Va_b = (const float*)d_in[8];
  const float* W_ih = (const float*)d_in[9];
  const float* W_hh = (const float*)d_in[10];
  const float* b_ih = (const float*)d_in[11];
  const float* b_hh = (const float*)d_in[12];
  const float* out_w = (const float*)d_in[13];
  const float* out_b = (const float*)d_in[14];

  float* out = (float*)d_out;
  float* lp_out   = out;                                   // [B,T,V]
  float* hT_out   = out + (size_t)B * T * V;               // [1,B,H]
  float* attn_out = hT_out + (size_t)B * H;                // [B,T,S]

  // workspace carve-out (256B aligned)
  char* w = (char*)d_ws;
  size_t off = 0;
  auto carve = [&](size_t bytes) -> char* {
    char* p = w + off;
    off = (off + bytes + 255) & ~(size_t)255;
    return p;
  };
  unsigned short* outw_b = (unsigned short*)carve((size_t)V * H * 2);
  unsigned short* Wih_b  = (unsigned short*)carve((size_t)G3 * H2 * 2);
  unsigned short* Whh_b  = (unsigned short*)carve((size_t)G3 * H * 2);
  unsigned short* Ua_b16 = (unsigned short*)carve((size_t)H * H * 2);
  unsigned short* keys_b = (unsigned short*)carve((size_t)B * S * H * 2);
  float*          Uk     = (float*)carve((size_t)B * S * H * 4);
  float*          hbuf   = (float*)carve((size_t)B * H * 4);
  unsigned short* hb     = (unsigned short*)carve((size_t)B * H * 2);
  unsigned short* xb     = (unsigned short*)carve((size_t)B * H2 * 2);
  float*          gx     = (float*)carve((size_t)B * G3 * 4);
  float*          gh     = (float*)carve((size_t)B * G3 * 4);
  float*          logits = (float*)carve((size_t)B * V * 4);
  int*            ids    = (int*)carve((size_t)B * 4);

  auto cvt = [&](const float* s, unsigned short* d, int n) {
    convert_bf16_kernel<<<(n + 255) / 256, 256, 0, stream>>>(s, d, n);
  };
  cvt(out_w, outw_b, V * H);
  cvt(W_ih, Wih_b, G3 * H2);
  cvt(W_hh, Whh_b, G3 * H);
  cvt(Ua_w, Ua_b16, H * H);
  cvt(encoder_outputs, keys_b, B * S * H);

  init_kernel<<<(B * H + 255) / 256, 256, 0, stream>>>(encoder_hidden, hbuf, hb, ids);
  uk_gemm_kernel<<<(B * S / 16) * (H / 16) / 8, 256, 0, stream>>>(keys_b, Ua_b16, Ua_b, Uk);

  for (int t = 0; t < T; ++t) {
    attn_kernel<<<B, 256, 0, stream>>>(hbuf, ids, embedding, encoder_outputs, Uk,
                                       Wa_w, Wa_b, Va_w, Va_b, xb, attn_out, t);
    gates_gemm_kernel<<<(G3 / 16) / 8, 256, 0, stream>>>(xb, hb, Wih_b, Whh_b,
                                                         b_ih, b_hh, gx, gh);
    gru_elem_kernel<<<(B * H) / 256, 256, 0, stream>>>(gx, gh, hbuf, hb, hT_out,
                                                       (t == T - 1) ? 1 : 0);
    logits_gemm_kernel<<<(V / 16) / 8, 256, 0, stream>>>(hb, outw_b, out_b, logits);
    reduce_kernel<<<B, 256, 0, stream>>>(logits, lp_out, ids, t);
  }
}
